// full_module_50379966382407
// MI455X (gfx1250) — matrix operand order repaired and checked by reading, not compiled, not run on hardware
//
#include <hip/hip_runtime.h>
#include <cstddef>

// ---------------------------------------------------------------------------
// WaveNet-ish 33-stage gated dilated conv stack + PinSage head for MI455X.
//
// Kernel 1: one workgroup per (stream s, batch b) line. Whole 16x4096 f16
//   activation line lives in LDS (x buffer + h buffer = 256KB of the 320KB
//   WGP LDS). All convs are v_wmma_f32_16x16x32_f16 with positions as N.
//   Dilated k=2 conv: K = tap*16+ch == 32 exactly; the two B-operand K-halves
//   (lanes 0-15 / 16-31) read the two tap positions -> dilation is just an
//   LDS address shift. Zero padding via address-selected zero guard row.
//   D->B transposes are done with __shfl_xor(.,16) half-wave exchanges (no
//   LDS round trip / DScnt drain). B operands are software-pipelined one tile
//   ahead. Next stage's weights are global_prefetch'ed during compute.
// Kernel 2: PinSage (16->16 x2 K-padded, concat -> 32->64) per 16-pos tile,
//   max over 3 streams in WMMA D registers, final 64->1 dot via VALU +
//   __shfl_xor cross-half reduction, sigmoid, store f32 output [B,L].
// ---------------------------------------------------------------------------

typedef __attribute__((ext_vector_type(16))) _Float16 v16h;
typedef __attribute__((ext_vector_type(8)))  _Float16 v8h;
typedef __attribute__((ext_vector_type(8)))  float    v8f;

union U4 { v8h  h; unsigned u[4]; };
union U8 { v16h h; unsigned u[8]; };

#define LSEQ   4096
#define NBATCH 8

// ----- kernel 1 geometry -----
#define T1     512                 // 16 waves of 32
#define W1CNT  (T1 / 32)
#define TPW    (256 / W1CNT)       // 16 N-tiles (of 16 positions) per wave

// kernel-1 dynamic LDS layout (bytes)
#define OFF_XBUF  0                         // 16ch x 4096 pos f16, [pos][ch]
#define OFF_HBUF  131072                    // same shape
#define OFF_WIMG  262144                    // 6 A-operand images * 512 halves
#define OFF_WBIAS 268288                    // 96 f32 biases
#define OFF_ZROW  268672                    // 32 f16 zeros: guard row
#define SMEM1     268736

// ----- kernel 2 geometry -----
#define T2     256                 // 8 waves; one 16-pos tile per wave
#define OFF2_WIMG  0               // 18 A-images * 512 halves = 18432
#define OFF2_BIAS  18432           // 3 * 96 f32 = 1152
#define OFF2_FW    19584           // 64 f32 final_w + 1 f32 final_b
#define SMEM2      19872

static __device__ __forceinline__ v8f wmma16(v16h a, v16h b, v8f c) {
  // D = A(16x32 f16) * B(32x16 f16) + C(16x16 f32)
  return __builtin_amdgcn_wmma_f32_16x16x32_f16(false, a, false, b, (short)0, c,
                                                false, false);
}

// Native-trans gating: v_exp_f32 + v_rcp_f32 instead of ocml tanhf expansion.
static __device__ __forceinline__ float fast_sigmoid(float x) {
  return __builtin_amdgcn_rcpf(1.0f + __expf(-x));
}
static __device__ __forceinline__ float fast_tanh(float x) {
  // tanh(x) = 1 - 2/(exp(2x)+1); saturates correctly as exp -> 0 or inf
  return 1.0f - 2.0f * __builtin_amdgcn_rcpf(__expf(2.0f * x) + 1.0f);
}

// ---------------------------------------------------------------------------
// Kernel 1: the 33 dilated stages for one (stream, batch) line.
// grid = (8, 3); block = 512 threads = 16 waves.
// ---------------------------------------------------------------------------
__global__ void __launch_bounds__(T1)
wavenet_stages_kernel(const float* __restrict__ xin,
                      const float* __restrict__ c1f_w, const float* __restrict__ c1f_b,
                      const float* __restrict__ c1p_w, const float* __restrict__ c1p_b,
                      const float* __restrict__ cf_w,  const float* __restrict__ cf_b,
                      const float* __restrict__ cg_w,  const float* __restrict__ cg_b,
                      const float* __restrict__ c2_w,  const float* __restrict__ c2_b,
                      float* __restrict__ skip) {
  extern __shared__ char smem[];
  _Float16* xbuf  = (_Float16*)(smem + OFF_XBUF);
  _Float16* hbuf  = (_Float16*)(smem + OFF_HBUF);
  _Float16* wimg  = (_Float16*)(smem + OFF_WIMG);
  float*    wbias = (float*)   (smem + OFF_WBIAS);
  _Float16* zrow  = (_Float16*)(smem + OFF_ZROW);

  const int bb   = blockIdx.x;          // batch
  const int s    = blockIdx.y;          // stream / submodule
  const int tid  = threadIdx.x;
  const int wave = tid >> 5;
  const int lane = tid & 31;
  const int hf   = lane >> 4;           // which K/M half of the wave
  const int n    = lane & 15;           // N (position-within-tile) index
  float* skipBase = skip + ((size_t)(s * NBATCH + bb) * LSEQ) * 16;

  if (tid < 32) zrow[tid] = (_Float16)0.0f;   // guard row (barrier below covers)

  for (int i = 0; i < 33; ++i) {
    const int d = 1 << (i % 11);

    // ---- stage weight prep: build WMMA A-operand register images in LDS ----
    for (int idx = tid; idx < 6 * 512; idx += T1) {
      const int tile = idx >> 9;
      const int ln   = (idx >> 4) & 31;
      const int j    = idx & 15;
      const int m    = ln & 15;
      // A image uses contiguous k (B matches): lanes 0-15 hold K 0-15, lanes 16-31 K 16-31
      const int k = j + ((ln >= 16) ? 16 : 0);
      float w = 0.0f;
      if (tile == 0) {                       // W1: 16x16, K padded to 32
        if (i > 0 && k < 16) w = c1p_w[((s * 32 + (i - 1)) * 16 + m) * 16 + k];
      } else if (tile <= 4) {                // Wf lo/hi, Wg lo/hi: K = tap*16+ch
        const float* wb = (tile <= 2) ? cf_w : cg_w;
        const int o = m + ((((tile - 1) & 1)) ? 16 : 0);
        const int c = k & 15, t = k >> 4;
        w = wb[(((s * 33 + i) * 32 + o) * 16 + c) * 2 + t];
      } else {                               // W2: 16x32
        w = c2_w[((s * 33 + i) * 16 + m) * 32 + k];
      }
      wimg[idx] = (_Float16)w;
    }
    if (tid < 96) {
      float v;
      if (tid < 16)      v = (i > 0) ? c1p_b[(s * 32 + (i - 1)) * 16 + tid] : 0.0f;
      else if (tid < 48) v = cf_b[(s * 33 + i) * 32 + (tid - 16)];
      else if (tid < 80) v = cg_b[(s * 33 + i) * 32 + (tid - 48)];
      else               v = c2_b[(s * 33 + i) * 16 + (tid - 80)];
      wbias[tid] = v;
    }
    // prefetch next stage's weights into cache while this stage computes
    if (i + 1 < 33) {
      const int ip = i + 1;
      if (tid < 64) {
        __builtin_prefetch(cf_w + (size_t)(s * 33 + ip) * 1024 + tid * 16, 0, 3);
        __builtin_prefetch(cg_w + (size_t)(s * 33 + ip) * 1024 + tid * 16, 0, 3);
      } else if (tid < 96) {
        __builtin_prefetch(c2_w + (size_t)(s * 33 + ip) * 512 + (tid - 64) * 16, 0, 3);
      } else if (tid < 112) {
        __builtin_prefetch(c1p_w + (size_t)(s * 32 + ip - 1) * 256 + (tid - 96) * 16, 0, 3);
      }
    }
    __syncthreads();

    // per-wave A operands (resident across the tile loop) + bias C-vectors
    const v16h* wv = (const v16h*)wimg;
    const v16h a_w1 = wv[lane];
    const v16h a_f0 = wv[32 + lane],  a_f1 = wv[64 + lane];
    const v16h a_g0 = wv[96 + lane],  a_g1 = wv[128 + lane];
    const v16h a_w2 = wv[160 + lane];
    v8f b1v, bf0, bf1, bg0, bg1, b2v;
#pragma unroll
    for (int r = 0; r < 8; ++r) {
      const int mb = r + hf * 8;          // D-layout M index for this lane/vgpr
      b1v[r] = wbias[mb];
      bf0[r] = wbias[16 + mb];  bf1[r] = wbias[32 + mb];
      bg0[r] = wbias[48 + mb];  bg1[r] = wbias[64 + mb];
      b2v[r] = wbias[80 + mb];
    }

    // ---- step 1: h = relu(W1 . x + b1) (pointwise) ----
    if (i == 0) {
      // 1 -> 16 first conv from the global input, plain VALU
      float w0[16], b0[16];
#pragma unroll
      for (int m = 0; m < 16; ++m) {
        w0[m] = c1f_w[s * 16 + m];
        b0[m] = c1f_b[s * 16 + m];
      }
      for (int pos = tid; pos < LSEQ; pos += T1) {
        const float xv = xin[((size_t)bb * LSEQ + pos) * 3 + s];
        v8h h0, h1;
#pragma unroll
        for (int m = 0; m < 8; ++m) {
          h0[m] = (_Float16)fmaxf(w0[m] * xv + b0[m], 0.0f);
          h1[m] = (_Float16)fmaxf(w0[m + 8] * xv + b0[m + 8], 0.0f);
        }
        *(v8h*)(hbuf + pos * 16)     = h0;
        *(v8h*)(hbuf + pos * 16 + 8) = h1;
      }
    } else {
      // software-pipelined: preload next tile's B operand during this WMMA
      const _Float16* xsrc = hf ? zrow : (xbuf + (wave * TPW * 16 + n) * 16);
      v16h xb = *(const v16h*)xsrc;
      for (int t = 0; t < TPW; ++t) {
        const int p0 = (wave * TPW + t) * 16;
        v16h xb_nxt = xb;
        if (t + 1 < TPW) {
          const _Float16* xs = hf ? zrow : (xbuf + (p0 + 16 + n) * 16);
          xb_nxt = *(const v16h*)xs;
        }
        v8f acc = wmma16(a_w1, xb, b1v);
        v8h hh;
#pragma unroll
        for (int r = 0; r < 8; ++r) hh[r] = (_Float16)fmaxf(acc[r], 0.0f);
        *(v8h*)(hbuf + (p0 + n) * 16 + hf * 8) = hh;
        xb = xb_nxt;
      }
    }
    __syncthreads();   // h complete before cross-position dilated reads

    // ---- step 2: gated dilated conv, z = relu(W2.(tanh*sig)), x = h + z ----
    const int dsh = hf ? (d - d / 2) : -(d / 2);   // tap shift per K-half
    {
      const int pos0 = wave * TPW * 16 + n + dsh;
      const _Float16* h0src =
          ((unsigned)pos0 < (unsigned)LSEQ) ? (hbuf + pos0 * 16) : zrow;
      v16h hbv = *(const v16h*)h0src;
      for (int t = 0; t < TPW; ++t) {
        const int p0 = (wave * TPW + t) * 16;
        v16h hbv_nxt = hbv;
        if (t + 1 < TPW) {
          const int posn = p0 + 16 + n + dsh;
          const _Float16* hs =
              ((unsigned)posn < (unsigned)LSEQ) ? (hbuf + posn * 16) : zrow;
          hbv_nxt = *(const v16h*)hs;
        }
        // residual h row for this tile (own positions) - issue early
        const v8h hv = *(const v8h*)(hbuf + (p0 + n) * 16 + hf * 8);
        v8f f0 = wmma16(a_f0, hbv, bf0);
        v8f f1 = wmma16(a_f1, hbv, bf1);
        v8f g0 = wmma16(a_g0, hbv, bg0);
        v8f g1 = wmma16(a_g1, hbv, bg1);
        // gate
        v8h z0h, z1h;
#pragma unroll
        for (int r = 0; r < 8; ++r) {
          z0h[r] = (_Float16)(fast_tanh(f0[r]) * fast_sigmoid(g0[r]));
          z1h[r] = (_Float16)(fast_tanh(f1[r]) * fast_sigmoid(g1[r]));
        }
        // D -> B transpose via half-wave exchange (no LDS, no DScnt drain):
        // B lane(n,half0) needs gated ch0-15, lane(n,half1) ch16-31.
        U4 uz0, uz1; uz0.h = z0h; uz1.h = z1h;
        unsigned oz0[4], oz1[4];
#pragma unroll
        for (int q = 0; q < 4; ++q) {
          oz0[q] = (unsigned)__shfl_xor((int)uz0.u[q], 16, 32);
          oz1[q] = (unsigned)__shfl_xor((int)uz1.u[q], 16, 32);
        }
        U8 ug;
#pragma unroll
        for (int q = 0; q < 4; ++q) {
          ug.u[q]     = hf ? oz1[q] : uz0.u[q];   // K 0-7  / 16-23
          ug.u[q + 4] = hf ? uz1.u[q] : oz0[q];   // K 8-15 / 24-31
        }
        v8f zc = wmma16(a_w2, ug.h, b2v);
        // residual x = h + z, skip += z (f32 in L2-resident global scratch)
        v8h xo;
        float zr[8];
#pragma unroll
        for (int r = 0; r < 8; ++r) {
          const float z = fmaxf(zc[r], 0.0f);
          zr[r] = z;
          xo[r] = (_Float16)((float)hv[r] + z);
        }
        *(v8h*)(xbuf + (p0 + n) * 16 + hf * 8) = xo;
        float* sk = skipBase + (size_t)(p0 + n) * 16 + hf * 8;
        if (i == 0) {
#pragma unroll
          for (int r = 0; r < 8; ++r) sk[r] = zr[r];
        } else {
#pragma unroll
          for (int r = 0; r < 8; ++r) sk[r] += zr[r];
        }
        hbv = hbv_nxt;
      }
    }
    __syncthreads();   // all dilated reads of h done before next stage rewrites it
  }
}

// ---------------------------------------------------------------------------
// Kernel 2: relu(skip) -> PinSage (per stream) -> max over streams -> 64->1
// grid = 256 blocks * 8 waves = 2048 tiles = 8 batch * 256 position tiles.
// ---------------------------------------------------------------------------
__global__ void __launch_bounds__(T2)
pinsage_head_kernel(const float* __restrict__ skip,
                    const float* __restrict__ n1w, const float* __restrict__ n1b,
                    const float* __restrict__ n2w, const float* __restrict__ n2b,
                    const float* __restrict__ upw, const float* __restrict__ upb,
                    const float* __restrict__ fw,  const float* __restrict__ fb,
                    float* __restrict__ out) {
  extern __shared__ char smem[];
  _Float16* wimg = (_Float16*)(smem + OFF2_WIMG);
  float*    blds = (float*)   (smem + OFF2_BIAS);
  float*    fwl  = (float*)   (smem + OFF2_FW);

  const int tid  = threadIdx.x;
  const int wave = tid >> 5;
  const int lane = tid & 31;
  const int hf   = lane >> 4;
  const int n    = lane & 15;

  // --- A-operand images: per stream {n1, n2, up0..up3} -> 18 tiles ---
  for (int idx = tid; idx < 18 * 512; idx += T2) {
    const int tt = idx >> 9;
    const int ln = (idx >> 4) & 31;
    const int j  = idx & 15;
    const int sS = tt / 6, q = tt % 6;
    const int m  = ln & 15;
    const int k  = j + ((ln >= 16) ? 16 : 0);
    float w = 0.0f;
    if (q == 0)      { if (k < 16) w = n1w[(sS * 16 + m) * 16 + k]; }
    else if (q == 1) { if (k < 16) w = n2w[(sS * 16 + m) * 16 + k]; }
    else             { const int o = (q - 2) * 16 + m;
                       w = upw[(sS * 64 + o) * 32 + k]; }   // cat order: [t | nb]
    wimg[idx] = (_Float16)w;
  }
  if (tid < 288) {
    const int sS = tid / 96, r = tid % 96;
    float v;
    if (r < 16)      v = n1b[sS * 16 + r];
    else if (r < 32) v = n2b[sS * 16 + (r - 16)];
    else             v = upb[sS * 64 + (r - 32)];
    blds[tid] = v;
  }
  if (tid < 64) fwl[tid] = fw[tid];
  if (tid == 64) fwl[64] = fb[0];
  __syncthreads();

  const int tileId = blockIdx.x * (T2 / 32) + wave;   // 0..2047
  const int bb = tileId >> 8;
  const int p0 = (tileId & 255) * 16;

  // load the three stream features as B operands (K 0-15 = ch, K 16-31 = pad)
  v16h tb[3];
#pragma unroll
  for (int s3 = 0; s3 < 3; ++s3) {
    v16h v = {};
    if (hf == 0) {
      const float* p = skip + ((size_t)(s3 * NBATCH + bb) * LSEQ + p0 + n) * 16;
#pragma unroll
      for (int c = 0; c < 16; ++c) v[c] = (_Float16)fmaxf(p[c], 0.0f);
    }
    tb[s3] = v;
  }

  const int TI[3] = {0, 1, 2};   // target stream per pinsage block
  const int N1[3] = {1, 0, 0};
  const int N2[3] = {2, 2, 1};
  const v16h* wv = (const v16h*)wimg;

  v8f om[4];
#pragma unroll
  for (int sS = 0; sS < 3; ++sS) {
    const v16h* ws = wv + sS * 6 * 32;
    v8f bn1, bn2;
#pragma unroll
    for (int r = 0; r < 8; ++r) {
      bn1[r] = blds[sS * 96 + r + hf * 8];
      bn2[r] = blds[sS * 96 + 16 + r + hf * 8];
    }
    v8f a1 = wmma16(ws[lane],      tb[N1[sS]], bn1);
    v8f a2 = wmma16(ws[32 + lane], tb[N2[sS]], bn2);
    v8h nbh;
#pragma unroll
    for (int r = 0; r < 8; ++r)
      nbh[r] = (_Float16)fmaxf(fmaxf(a1[r], 0.0f), fmaxf(a2[r], 0.0f));
    // cat B operand: K0-15 = t channels (from tb), K16-31 = nb via half-wave
    // exchange of the packed D-layout nb values (no LDS round trip).
    U4 unb; unb.h = nbh;
    unsigned onb[4];
#pragma unroll
    for (int q = 0; q < 4; ++q)
      onb[q] = (unsigned)__shfl_xor((int)unb.u[q], 16, 32);
    U8 ut; ut.h = tb[TI[sS]];
    U8 uc;
#pragma unroll
    for (int q = 0; q < 4; ++q) {
      uc.u[q]     = hf ? onb[q]   : ut.u[q];
      uc.u[q + 4] = hf ? unb.u[q] : ut.u[q + 4];
    }
    const v16h catb = uc.h;
#pragma unroll
    for (int mc = 0; mc < 4; ++mc) {
      v8f bu;
#pragma unroll
      for (int r = 0; r < 8; ++r) bu[r] = blds[sS * 96 + 32 + mc * 16 + r + hf * 8];
      v8f u = wmma16(ws[(2 + mc) * 32 + lane], catb, bu);
#pragma unroll
      for (int r = 0; r < 8; ++r) u[r] = fmaxf(u[r], 0.0f);
      if (sS == 0) om[mc] = u;
      else {
#pragma unroll
        for (int r = 0; r < 8; ++r) om[mc][r] = fmaxf(om[mc][r], u[r]);
      }
    }
  }

  // final 64->1 dot across the D-layout M indices + cross-half shuffle reduce
  float partial = 0.0f;
#pragma unroll
  for (int mc = 0; mc < 4; ++mc)
#pragma unroll
    for (int r = 0; r < 8; ++r)
      partial += fwl[mc * 16 + r + hf * 8] * om[mc][r];
  const float tot = partial + __shfl_xor(partial, 16, 32);
  if (hf == 0)
    out[(size_t)bb * LSEQ + p0 + n] =
        __builtin_amdgcn_rcpf(1.0f + __expf(-(tot + fwl[64])));
}

// ---------------------------------------------------------------------------
extern "C" void kernel_launch(void* const* d_in, const int* in_sizes, int n_in,
                              void* d_out, int out_size, void* d_ws, size_t ws_size,
                              hipStream_t stream) {
  const float* xin   = (const float*)d_in[0];
  const float* c1f_w = (const float*)d_in[1];
  const float* c1f_b = (const float*)d_in[2];
  const float* c1p_w = (const float*)d_in[3];
  const float* c1p_b = (const float*)d_in[4];
  const float* cf_w  = (const float*)d_in[5];
  const float* cf_b  = (const float*)d_in[6];
  const float* cg_w  = (const float*)d_in[7];
  const float* cg_b  = (const float*)d_in[8];
  const float* c2_w  = (const float*)d_in[9];
  const float* c2_b  = (const float*)d_in[10];
  const float* n1w   = (const float*)d_in[11];
  const float* n1b   = (const float*)d_in[12];
  const float* n2w   = (const float*)d_in[13];
  const float* n2b   = (const float*)d_in[14];
  const float* upw   = (const float*)d_in[15];
  const float* upb   = (const float*)d_in[16];
  const float* fw    = (const float*)d_in[17];
  const float* fb    = (const float*)d_in[18];
  float* out  = (float*)d_out;
  float* skip = (float*)d_ws;   // 3*8*4096*16 f32 = 6.3 MB (L2-resident)

  // kernel 1 needs 263 KB dynamic LDS (of the 320 KB WGP pool)
  (void)hipFuncSetAttribute((const void*)wavenet_stages_kernel,
                            hipFuncAttributeMaxDynamicSharedMemorySize, SMEM1);
  (void)hipFuncSetAttribute((const void*)pinsage_head_kernel,
                            hipFuncAttributeMaxDynamicSharedMemorySize, SMEM2);

  dim3 g1(NBATCH, 3);
  wavenet_stages_kernel<<<g1, T1, SMEM1, stream>>>(
      xin, c1f_w, c1f_b, c1p_w, c1p_b, cf_w, cf_b, cg_w, cg_b, c2_w, c2_b, skip);

  pinsage_head_kernel<<<256, T2, SMEM2, stream>>>(
      skip, n1w, n1b, n2w, n2b, upw, upb, fw, fb, out);
}
